// ModifiedEncoderLayer_1675037245816
// MI455X (gfx1250) — matrix operand order repaired and checked by reading, not compiled, not run on hardware
//
#include <hip/hip_runtime.h>
#include <math.h>

// ---------------- problem constants ----------------
constexpr int NT   = 384;   // tokens
constexpr int HIDC = 256;
constexpr int REPC = 512;
constexpr int HC   = 8;
constexpr int DC   = 64;

typedef __bf16 bf16;
typedef __attribute__((ext_vector_type(16))) __bf16 v16bf;
typedef __attribute__((ext_vector_type(8)))  float  v8f;

constexpr int LDSP = 40;   // padded LDS row stride in bf16 elems (80B, 16B aligned)

// ---- load 8 elems as packed bf16x8 (16B). Independent temps let the
// compiler clause all global loads before converting/storing. ----
__device__ __forceinline__ uint4 ld8pack(const float* __restrict__ s) {
  const float4* p = (const float4*)s;
  float4 q0 = p[0], q1 = p[1];
  union { bf16 h[8]; uint4 u; } r;
  r.h[0]=(bf16)q0.x; r.h[1]=(bf16)q0.y; r.h[2]=(bf16)q0.z; r.h[3]=(bf16)q0.w;
  r.h[4]=(bf16)q1.x; r.h[5]=(bf16)q1.y; r.h[6]=(bf16)q1.z; r.h[7]=(bf16)q1.w;
  return r.u;
}
__device__ __forceinline__ uint4 ld8pack(const bf16* __restrict__ s) {
  return *(const uint4*)s;
}

// ---- fragment loads per CDNA5 16-bit WMMA VGPR layouts (wave32) ----
// A (16x32): lanes 0-15 row M=lane, halves 0..7 = K 0..7, 8..15 = K 16..23
//            lanes 16-31 row M=lane-16, halves = K 8..15 / 24..31
__device__ __forceinline__ v16bf load_frag_a(const bf16* sA, int rowBase, int lane) {
  const bf16* base = sA + (rowBase + (lane & 15)) * LDSP;
  const int hi = (lane < 16) ? 0 : 8;
  v16bf a;
#pragma unroll
  for (int e = 0; e < 8; ++e) a[e]     = base[hi + e];
#pragma unroll
  for (int e = 0; e < 8; ++e) a[8 + e] = base[16 + hi + e];
  return a;
}
// B (32x16): lane n = lane&15; halves 0..7 = K 0..7 / 8..15, 8..15 = K 16..23 / 24..31 (lanes<16 / >=16)
__device__ __forceinline__ v16bf load_frag_b(const bf16* sB, int nBase, int lane) {
  const bf16* base = sB + (nBase + (lane & 15)) * LDSP;
  const int hi = (lane < 16) ? 0 : 8;
  v16bf b;
#pragma unroll
  for (int e = 0; e < 8; ++e) b[e]     = base[hi + e];
#pragma unroll
  for (int e = 0; e < 8; ++e) b[8 + e] = base[16 + hi + e];
  return b;
}

// ---------------- generic GEMM: out = act(A(MxK) * W(NnxK)^T + bias) -------
// Tile 128x128, 8 waves (4M x 2N), wave = 2x4 16x16x32 bf16 WMMA frags.
// Double-buffered LDS. Optional fused pair-logit epilogue (FUSE_LP):
// a wave's 64 columns == one head, lp[h][j][i] = sum_d relu(p)[ij, h*64+d]*w_p[h][d].
template <typename AT, bool FUSE_LP>
__global__ __launch_bounds__(256) void gemm_wmma(
    const AT* __restrict__ A, const float* __restrict__ W,
    const float* __restrict__ bias, float* __restrict__ outF,
    bf16* __restrict__ outB, float* __restrict__ lp,
    const float* __restrict__ attn_w, int M, int Nn, int K, int relu) {
  __shared__ bf16 sA[2][128 * LDSP];
  __shared__ bf16 sB[2][128 * LDSP];

  const int tid   = threadIdx.x;
  const int lane  = tid & 31;
  const int wave  = tid >> 5;       // 0..7
  const int waveM = wave >> 1;      // 0..3 -> 32 rows each
  const int waveN = wave & 1;       // 0..1 -> 64 cols each
  const int nBase = blockIdx.x * 128;   // N-tiles fastest -> A reused in L2
  const int mBase = blockIdx.y * 128;

  const int r0 = tid >> 2,         k0c = (tid & 3) << 3;          // chunk 0
  const int r1 = (tid + 256) >> 2, k1c = ((tid + 256) & 3) << 3;  // chunk 1

  v8f acc[2][4] = {};

#define STAGE(buf, kk)                                                         \
  do {                                                                         \
    uint4 ta0 = ld8pack(A + (size_t)(mBase + r0) * K + (kk) + k0c);            \
    uint4 ta1 = ld8pack(A + (size_t)(mBase + r1) * K + (kk) + k1c);            \
    uint4 tb0 = ld8pack(W + (size_t)(nBase + r0) * K + (kk) + k0c);            \
    uint4 tb1 = ld8pack(W + (size_t)(nBase + r1) * K + (kk) + k1c);            \
    *(uint4*)&sA[buf][r0 * LDSP + k0c] = ta0;                                  \
    *(uint4*)&sA[buf][r1 * LDSP + k1c] = ta1;                                  \
    *(uint4*)&sB[buf][r0 * LDSP + k0c] = tb0;                                  \
    *(uint4*)&sB[buf][r1 * LDSP + k1c] = tb1;                                  \
  } while (0)

  STAGE(0, 0);
  __syncthreads();
  const int ksteps = K >> 5;
  for (int s = 0; s < ksteps; ++s) {
    const int buf = s & 1;
    if (s + 1 < ksteps) STAGE(buf ^ 1, (s + 1) << 5);   // overlap with WMMA
    v16bf af[2], bfr[4];
#pragma unroll
    for (int m = 0; m < 2; ++m) af[m]  = load_frag_a(sA[buf], waveM * 32 + m * 16, lane);
#pragma unroll
    for (int n = 0; n < 4; ++n) bfr[n] = load_frag_b(sB[buf], waveN * 64 + n * 16, lane);
#pragma unroll
    for (int m = 0; m < 2; ++m)
#pragma unroll
      for (int n = 0; n < 4; ++n)
        acc[m][n] = __builtin_amdgcn_wmma_f32_16x16x32_bf16(
            false, af[m], false, bfr[n], (short)0, acc[m][n], false, false);
    __syncthreads();
  }
#undef STAGE

  // epilogue. C/D layout: VGPR r: lanes0-15 M=r, lanes16-31 M=r+8; N=lane&15
  const int rowW = mBase + waveM * 32 + ((lane < 16) ? 0 : 8);
  const int colW = nBase + waveN * 64 + (lane & 15);
#pragma unroll
  for (int m = 0; m < 2; ++m) {
#pragma unroll
    for (int n = 0; n < 4; ++n) {
      int col = colW + n * 16;
      float bv = bias[col];
#pragma unroll
      for (int r = 0; r < 8; ++r) {
        float v = acc[m][n][r] + bv;
        if (relu) v = fmaxf(v, 0.f);
        acc[m][n][r] = v;
        size_t idx = (size_t)(rowW + m * 16 + r) * Nn + col;
        if (outF) outF[idx] = v;
        if (outB) outB[idx] = (bf16)v;
      }
    }
  }

  if (FUSE_LP) {
    // head for this wave; d = n*16 + (lane&15)
    const int h = (nBase >> 6) + waveN;
    const float* wp = attn_w + h * 3 * DC + 2 * DC;
#pragma unroll
    for (int m = 0; m < 2; ++m) {
      float part[8];
#pragma unroll
      for (int r = 0; r < 8; ++r) part[r] = 0.f;
#pragma unroll
      for (int n = 0; n < 4; ++n) {
        float w_ = wp[n * 16 + (lane & 15)];
#pragma unroll
        for (int r = 0; r < 8; ++r) part[r] += acc[m][n][r] * w_;
      }
#pragma unroll
      for (int r = 0; r < 8; ++r) {
        part[r] += __shfl_xor(part[r], 1);
        part[r] += __shfl_xor(part[r], 2);
        part[r] += __shfl_xor(part[r], 4);
        part[r] += __shfl_xor(part[r], 8);
      }
      if ((lane & 15) == 0) {
#pragma unroll
        for (int r = 0; r < 8; ++r) {
          int row = rowW + m * 16 + r;      // flat ij
          int i = row / NT, j = row % NT;
          lp[((size_t)h * NT + j) * NT + i] = part[r];
        }
      }
    }
  }
}

// ---------------- li/lj: per-token per-head logit contributions ------------
__global__ __launch_bounds__(256) void lilj_kernel(
    const bf16* __restrict__ u_bf, const float* __restrict__ attn_w,
    float* __restrict__ li, float* __restrict__ lj) {
  const int i = blockIdx.x;
  const int h = threadIdx.x >> 5, l = threadIdx.x & 31;
  float a = 0.f, b = 0.f;
  for (int d = l; d < DC; d += 32) {
    float uv = (float)u_bf[(size_t)i * REPC + h * DC + d];
    a += uv * attn_w[h * 3 * DC + d];
    b += uv * attn_w[h * 3 * DC + DC + d];
  }
  for (int off = 16; off > 0; off >>= 1) {
    a += __shfl_down(a, off);
    b += __shfl_down(b, off);
  }
  if (l == 0) { li[h * NT + i] = a; lj[h * NT + i] = b; }
}

// ------- softmax over i (contiguous in [h][j][i]); folds li/lj/attn_b ------
__global__ __launch_bounds__(128) void softmax_kernel(
    float* __restrict__ logits, const float* __restrict__ li,
    const float* __restrict__ lj, const float* __restrict__ attn_b) {
  const int blk = blockIdx.x;               // h*NT + j
  const int h = blk / NT, j = blk % NT;
  float* row = logits + (size_t)blk * NT;
  const float* lih = li + (size_t)h * NT;
  const float addc = lj[(size_t)h * NT + j] + attn_b[h];
  const int tid = threadIdx.x;
  __shared__ float red[128];
  float m = -1e30f;
  for (int i = tid; i < NT; i += 128) m = fmaxf(m, row[i] + lih[i] + addc);
  red[tid] = m; __syncthreads();
  for (int s = 64; s > 0; s >>= 1) { if (tid < s) red[tid] = fmaxf(red[tid], red[tid + s]); __syncthreads(); }
  m = red[0]; __syncthreads();
  float sum = 0.f;
  for (int i = tid; i < NT; i += 128) {
    float e = __expf(row[i] + lih[i] + addc - m);
    row[i] = e; sum += e;
  }
  red[tid] = sum; __syncthreads();
  for (int s = 64; s > 0; s >>= 1) { if (tid < s) red[tid] += red[tid + s]; __syncthreads(); }
  float inv = 1.f / red[0];
  for (int i = tid; i < NT; i += 128) row[i] *= inv;
}

// ------- weighted aggregation + per-head 64x64 matvec (linearity trick) ----
__global__ __launch_bounds__(256) void agg_kernel(
    const float* __restrict__ wsoft,  // [h][j][i]
    const bf16* __restrict__ u_bf, const bf16* __restrict__ p_bf,
    const float* __restrict__ msg_w, const float* __restrict__ msg_b,
    bf16* __restrict__ agg_bf) {
  const int blk = blockIdx.x;           // h*NT + j
  const int h = blk / NT, j = blk % NT;
  const int d = threadIdx.x & 63, ig = threadIdx.x >> 6;  // 4 i-groups
  __shared__ float sA[4][DC];
  __shared__ float sO[4][DC];
  const float* wrow = wsoft + ((size_t)h * NT + j) * NT;
  float acc = 0.f;
  for (int i = ig; i < NT; i += 4) {
    float wv = wrow[i];
    float uv = (float)u_bf[(size_t)i * REPC + h * DC + d];
    float pv = (float)p_bf[((size_t)i * NT + j) * REPC + h * DC + d];
    acc += wv * uv * pv;
  }
  sA[ig][d] = acc;
  __syncthreads();
  if (ig == 0) sA[0][d] = sA[0][d] + sA[1][d] + sA[2][d] + sA[3][d];
  __syncthreads();
  float out = 0.f;
  const int e = d, dg = ig;
#pragma unroll 4
  for (int dd = dg * 16; dd < dg * 16 + 16; ++dd)
    out += sA[0][dd] * msg_w[((size_t)h * DC + dd) * DC + e];
  sO[dg][e] = out;
  __syncthreads();
  if (dg == 0) {
    float v = sO[0][e] + sO[1][e] + sO[2][e] + sO[3][e] + msg_b[h * DC + e];
    agg_bf[(size_t)j * REPC + h * DC + e] = (bf16)fmaxf(v, 0.f);
  }
}

// ---------------- layernorm over HID=256 (block = one row) ----------------
__global__ __launch_bounds__(HIDC) void ln_kernel(
    const float* __restrict__ a, const float* __restrict__ b,
    const float* __restrict__ g, const float* __restrict__ beta,
    float* __restrict__ outF, bf16* __restrict__ outB) {
  const int row = blockIdx.x, tid = threadIdx.x;
  __shared__ float red[HIDC];
  float v = a[(size_t)row * HIDC + tid] + b[(size_t)row * HIDC + tid];
  red[tid] = v; __syncthreads();
  for (int s = HIDC / 2; s > 0; s >>= 1) { if (tid < s) red[tid] += red[tid + s]; __syncthreads(); }
  float mu = red[0] * (1.f / HIDC);
  __syncthreads();
  float dv = v - mu;
  red[tid] = dv * dv; __syncthreads();
  for (int s = HIDC / 2; s > 0; s >>= 1) { if (tid < s) red[tid] += red[tid + s]; __syncthreads(); }
  float var = red[0] * (1.f / HIDC);
  float yv = dv * rsqrtf(var + 1e-5f) * g[tid] + beta[tid];
  if (outF) outF[(size_t)row * HIDC + tid] = yv;
  if (outB) outB[(size_t)row * HIDC + tid] = (bf16)yv;
}

// ---------------- driver ----------------------------------------------------
extern "C" void kernel_launch(void* const* d_in, const int* in_sizes, int n_in,
                              void* d_out, int out_size, void* d_ws, size_t ws_size,
                              hipStream_t stream) {
  (void)in_sizes; (void)n_in; (void)out_size; (void)ws_size;
  const float* x       = (const float*)d_in[0];
  const float* y       = (const float*)d_in[1];
  const float* unary_w = (const float*)d_in[2];
  const float* unary_b = (const float*)d_in[3];
  const float* pair_w  = (const float*)d_in[4];
  const float* pair_b  = (const float*)d_in[5];
  const float* attn_w  = (const float*)d_in[6];
  const float* attn_b  = (const float*)d_in[7];
  const float* msg_w   = (const float*)d_in[8];
  const float* msg_b   = (const float*)d_in[9];
  const float* agg_w   = (const float*)d_in[10];
  const float* agg_b   = (const float*)d_in[11];
  const float* norm_g  = (const float*)d_in[12];
  const float* norm_b  = (const float*)d_in[13];
  const float* ffn_w1  = (const float*)d_in[14];
  const float* ffn_b1  = (const float*)d_in[15];
  const float* ffn_w2  = (const float*)d_in[16];
  const float* ffn_w2b = (const float*)d_in[17];
  const float* ffn_g   = (const float*)d_in[18];
  const float* ffn_bt  = (const float*)d_in[19];
  float* out = (float*)d_out;

  // workspace carve-up (256B aligned)
  char* ws = (char*)d_ws;
  size_t off = 0;
  auto alloc = [&](size_t bytes) { void* p = ws + off; off += (bytes + 255) & ~(size_t)255; return p; };
  bf16*  p_bf   = (bf16*) alloc((size_t)NT * NT * REPC * 2);   // 151 MB
  bf16*  u_bf   = (bf16*) alloc((size_t)NT * REPC * 2);
  float* li     = (float*)alloc((size_t)HC * NT * 4);
  float* lj     = (float*)alloc((size_t)HC * NT * 4);
  float* logits = (float*)alloc((size_t)HC * NT * NT * 4);     // lp -> weights in place
  bf16*  agg_bf = (bf16*) alloc((size_t)NT * REPC * 2);
  float* z      = (float*)alloc((size_t)NT * HIDC * 4);
  float* xn     = (float*)alloc((size_t)NT * HIDC * 4);
  bf16*  xn_bf  = (bf16*) alloc((size_t)NT * HIDC * 2);
  bf16*  h1_bf  = (bf16*) alloc((size_t)NT * 4 * HIDC * 2);
  float* f2     = (float*)alloc((size_t)NT * HIDC * 4);

  // 1) u = relu(x @ unary_w^T + unary_b)   [384 x 512, K=256]
  gemm_wmma<float, false><<<dim3(REPC / 128, NT / 128), 256, 0, stream>>>(
      x, unary_w, unary_b, nullptr, u_bf, nullptr, nullptr, NT, REPC, HIDC, 1);
  // 2) li/lj
  lilj_kernel<<<NT, 256, 0, stream>>>(u_bf, attn_w, li, lj);
  // 3) p = relu(y @ pair_w^T + pair_b) + fused pair logits  [(384*384) x 512]
  gemm_wmma<float, true><<<dim3(REPC / 128, (NT * NT) / 128), 256, 0, stream>>>(
      y, pair_w, pair_b, nullptr, p_bf, logits, attn_w, NT * NT, REPC, REPC, 1);
  // 4) softmax over i (adds li+lj+attn_b, normalizes in place)
  softmax_kernel<<<HC * NT, 128, 0, stream>>>(logits, li, lj, attn_b);
  // 5) weighted aggregation + per-head matvec -> relu -> agg_bf [384 x 512]
  agg_kernel<<<HC * NT, 256, 0, stream>>>(logits, u_bf, p_bf, msg_w, msg_b, agg_bf);
  // 6) z = agg @ agg_w^T + agg_b           [384 x 256, K=512]
  gemm_wmma<bf16, false><<<dim3(HIDC / 128, NT / 128), 256, 0, stream>>>(
      agg_bf, agg_w, agg_b, z, nullptr, nullptr, nullptr, NT, HIDC, REPC, 0);
  // 7) xn = LN(x + z)
  ln_kernel<<<NT, HIDC, 0, stream>>>(x, z, norm_g, norm_b, xn, xn_bf);
  // 8) h1 = relu(xn @ ffn_w1^T + ffn_b1)   [384 x 1024, K=256]
  gemm_wmma<bf16, false><<<dim3((4 * HIDC) / 128, NT / 128), 256, 0, stream>>>(
      xn_bf, ffn_w1, ffn_b1, nullptr, h1_bf, nullptr, nullptr, NT, 4 * HIDC, HIDC, 1);
  // 9) f2 = h1 @ ffn_w2^T + ffn_b2         [384 x 256, K=1024]
  gemm_wmma<bf16, false><<<dim3(HIDC / 128, NT / 128), 256, 0, stream>>>(
      h1_bf, ffn_w2, ffn_w2b, f2, nullptr, nullptr, nullptr, NT, HIDC, 4 * HIDC, 0);
  // 10) out = LN(xn + f2)
  ln_kernel<<<NT, HIDC, 0, stream>>>(xn, f2, ffn_g, ffn_bt, out, nullptr);
}
